// SoftmaxAttention_68736656605627
// MI455X (gfx1250) — matrix operand order repaired and checked by reading, not compiled, not run on hardware
//
#include <hip/hip_runtime.h>

// ---------------------------------------------------------------------------
// Fused masked cross-attention (ESIM-style), MI455X / gfx1250, wave32 WMMA.
// B=64, Lc=Lr=512, D=768, fp32 in/out.  Memory-bound (~400MB compulsory HBM
// traffic @ 23.3 TB/s ~= 17us); sim never touches HBM -- recomputed per
// direction, lives in LDS per 16-row query tile.  GEMMs use
// v_wmma_f32_16x16x32_f16 (f16 operands, f32 accumulate); softmax is f32 with
// reference-exact  w/(w.sum+1e-13)  semantics.  LDS rows are padded so
// fragment gathers hit distinct banks (odd dword stride) instead of
// the 16-way conflicts a 256B-multiple stride would cause.  Mask staging uses
// the CDNA5 async global->LDS path (ASYNCcnt).
// ---------------------------------------------------------------------------

typedef __attribute__((ext_vector_type(16))) _Float16 v16h;
typedef __attribute__((ext_vector_type(8)))  float    v8f;

union AFrag { v16h v; _Float16 h[16]; };

#define NB    64
#define LL    512
#define DD    768
#define TQ    16        // query rows per workgroup
#define NTHR  128       // 4 waves (wave32)
#define DCHUNK 384      // D staged in two chunks for the sim GEMM

// padded LDS row strides (halfs) -> dword strides 385, 193, 257, 17 (all odd)
#define QH_STR  770
#define KS_STR  386
#define WH_STR  514
#define VT_STR  34

#define STAGE_ELEMS 26112   // max(64*KS_STR=24704, 768*VT_STR=26112)

struct Smem {
  _Float16 Qh[TQ * QH_STR];     // query tile, f16              24640 B
  float    sim[TQ * LL];        // raw sim tile, f32            32768 B
  float    kmask[LL];           // key mask as float             2048 B
  _Float16 Wh[TQ * WH_STR];     // softmax weights, f16         16448 B
  float    redmax[TQ * 8];      // per-row reduction scratch      512 B
  float    redn[TQ * 8];        //                                512 B
  float    redz[TQ * 8];        //                                512 B
  float    qmask[TQ];           // query-row mask                  64 B
  _Float16 stage[STAGE_ELEMS];  // key / transposed-value stage 52224 B
};                              // total ~127 KB LDS

__global__ __launch_bounds__(NTHR, 1)
void esim_attn_kernel(const float* __restrict__ ctx,
                      const float* __restrict__ rsp,
                      const int*   __restrict__ cmask,
                      const int*   __restrict__ rmask,
                      float*       __restrict__ out)
{
  __shared__ Smem sm;

  const int qt  = blockIdx.x;          // query tile   0..31
  const int b   = blockIdx.y;          // batch        0..63
  const int dir = blockIdx.z;          // 0: attend_c, 1: attend_r

  const float* Q  = (dir == 0) ? ctx   : rsp;
  const float* V  = (dir == 0) ? rsp   : ctx;    // keys == values
  const int*   qm = (dir == 0) ? cmask : rmask;
  const int*   km = (dir == 0) ? rmask : cmask;
  float* O = out + (size_t)dir * NB * LL * DD;

  const int t     = threadIdx.x;
  const int lane  = t & 31;
  const int w     = t >> 5;
  const int qbase = qt * TQ;
  const size_t bOff = (size_t)b * LL * DD;

  // ---- Phase 0a: async-copy key mask ints straight into LDS (ASYNCcnt) ----
  // stage buffer is free until phase 1; park the raw ints in its first 2KB.
  {
    int* mstage = (int*)sm.stage;
    for (int i = t; i < LL; i += NTHR) {
      const unsigned ldsOff = (unsigned)(unsigned long long)(&mstage[i]);
      const unsigned long long ga = (unsigned long long)(const void*)&km[b * LL + i];
      asm volatile("global_load_async_to_lds_b32 %0, %1, off"
                   :: "v"(ldsOff), "v"(ga) : "memory");
    }
  }

  // ---- Phase 0b: stage query tile (f32 -> f16) while masks fly ------------
  for (int i = t; i < TQ * DD; i += NTHR) {
    const int r = i / DD, c = i - r * DD;
    sm.Qh[r * QH_STR + c] = (_Float16)Q[bOff + (size_t)(qbase + r) * DD + c];
  }
  if (t < TQ) sm.qmask[t] = (float)qm[b * LL + qbase + t];

  asm volatile("s_wait_asynccnt 0x0" ::: "memory");
  __syncthreads();
  {
    const int* mstage = (const int*)sm.stage;
    for (int i = t; i < LL; i += NTHR) sm.kmask[i] = (float)mstage[i];
  }
  // phase 1 re-syncs before overwriting stage

  const int fm = lane & 15;   // A: M-row / B: N-col owned by this lane
  const int fg = lane >> 4;   // lane group

  // ---- Phase 1: sim tile = Q(16xD) * K^T, 64 key rows per supertile --------
  for (int kt = 0; kt < LL / 64; ++kt) {
    v8f acc = 0.0f;
    for (int dc = 0; dc < 2; ++dc) {
      __syncthreads();                          // protect stage buffer reuse
      const int dbase = dc * DCHUNK;
      for (int i = t; i < 64 * DCHUNK; i += NTHR) {
        const int r = i / DCHUNK, c = i - r * DCHUNK;
        sm.stage[r * KS_STR + c] =
            (_Float16)V[bOff + (size_t)(kt * 64 + r) * DD + dbase + c];
      }
      __syncthreads();
      #pragma unroll
      for (int kk = 0; kk < DCHUNK / 32; ++kk) {
        AFrag a, bf;
        #pragma unroll
        for (int e = 0; e < 16; ++e) {
          const int ka = fg * 16 + e;                         // A 16x32 layout
          a.h[e]  = sm.Qh[fm * QH_STR + dbase + kk * 32 + ka];
          const int kb = fg * 16 + e;                         // B 32x16 layout
          bf.h[e] = sm.stage[(w * 16 + fm) * KS_STR + kk * 32 + kb];
        }
        acc = __builtin_amdgcn_wmma_f32_16x16x32_f16(
                false, a.v, false, bf.v, (short)0, acc, false, false);
      }
    }
    // scatter 16x16 block into the LDS sim tile (raw logits)
    const int keyl = kt * 64 + w * 16 + fm;
    #pragma unroll
    for (int r = 0; r < 8; ++r)
      sm.sim[(fg * 8 + r) * LL + keyl] = acc[r];
  }
  __syncthreads();

  // ---- Phase 2: masked softmax, reference-exact EPS handling ---------------
  // ref: s = softmax(l*m); w = s*m; out = w/(w.sum+EPS)  ==  num/(N + EPS*Z)
  // num_i = m_i*exp(l_i m_i - M), N = sum(num), Z = sum(exp(l_j m_j - M))
  {
    const int row = t >> 3;          // 0..15
    const int sub = t & 7;           // 0..7, 64 keys each
    float mx = -3.4e38f;
    for (int k = sub * 64; k < sub * 64 + 64; ++k)
      mx = fmaxf(mx, sm.sim[row * LL + k] * sm.kmask[k]);
    sm.redmax[row * 8 + sub] = mx;
    __syncthreads();
    float M = sm.redmax[row * 8];
    #pragma unroll
    for (int j = 1; j < 8; ++j) M = fmaxf(M, sm.redmax[row * 8 + j]);

    float sn = 0.f, sz = 0.f;
    for (int k = sub * 64; k < sub * 64 + 64; ++k) {
      const float mk = sm.kmask[k];
      const float e  = __expf(sm.sim[row * LL + k] * mk - M);
      sz += e;
      sn += mk * e;
    }
    sm.redn[row * 8 + sub] = sn;
    sm.redz[row * 8 + sub] = sz;
    __syncthreads();
    float N = 0.f, Z = 0.f;
    #pragma unroll
    for (int j = 0; j < 8; ++j) { N += sm.redn[row * 8 + j]; Z += sm.redz[row * 8 + j]; }
    const float inv = 1.0f / (N + 1e-13f * Z);
    for (int k = sub * 64; k < sub * 64 + 64; ++k) {
      const float mk = sm.kmask[k];
      sm.Wh[row * WH_STR + k] =
          (_Float16)(mk * __expf(sm.sim[row * LL + k] * mk - M) * inv);
    }
  }
  __syncthreads();

  // ---- Phase 3: attend = W(16x512) * V(512x768) ----------------------------
  v8f acc3[12];
  #pragma unroll
  for (int j = 0; j < 12; ++j) acc3[j] = 0.0f;

  for (int ks = 0; ks < LL / 32; ++ks) {
    __syncthreads();
    // stage 32 value rows transposed: stage[c*VT_STR + r] (c = feature, r = key)
    for (int i = t; i < 32 * DD; i += NTHR) {
      const int r = i / DD, c = i - r * DD;
      if (ks + 1 < LL / 32)
        __builtin_prefetch(&V[bOff + (size_t)((ks + 1) * 32 + r) * DD + c], 0, 1);
      sm.stage[c * VT_STR + r] = (_Float16)V[bOff + (size_t)(ks * 32 + r) * DD + c];
    }
    __syncthreads();

    AFrag a;
    #pragma unroll
    for (int e = 0; e < 16; ++e) {
      const int ka = fg * 16 + e;
      a.h[e] = sm.Wh[fm * WH_STR + ks * 32 + ka];
    }
    #pragma unroll
    for (int j = 0; j < 12; ++j) {
      const int nb = (w * 12 + j) * 16;
      AFrag bf;
      #pragma unroll
      for (int e = 0; e < 16; ++e) {
        const int kb = fg * 16 + e;
        bf.h[e] = sm.stage[(nb + fm) * VT_STR + kb];
      }
      acc3[j] = __builtin_amdgcn_wmma_f32_16x16x32_f16(
                  false, a.v, false, bf.v, (short)0, acc3[j], false, false);
    }
  }

  // ---- store, masked by query-row mask (coalesced over the 16 lane columns)
  #pragma unroll
  for (int j = 0; j < 12; ++j) {
    const int nb = (w * 12 + j) * 16;
    #pragma unroll
    for (int r = 0; r < 8; ++r) {
      const int mrow = fg * 8 + r;
      O[bOff + (size_t)(qbase + mrow) * DD + nb + fm] = acc3[j][r] * sm.qmask[mrow];
    }
  }
}

extern "C" void kernel_launch(void* const* d_in, const int* in_sizes, int n_in,
                              void* d_out, int out_size, void* d_ws, size_t ws_size,
                              hipStream_t stream) {
  const float* ctx = (const float*)d_in[0];   // (64, 512, 768) f32
  const float* rsp = (const float*)d_in[1];   // (64, 512, 768) f32
  const int*   cm  = (const int*)d_in[2];     // (64, 512) i32
  const int*   rm  = (const int*)d_in[3];     // (64, 512) i32
  float* out = (float*)d_out;                 // attend_c ++ attend_r, f32

  dim3 grid(LL / TQ, NB, 2);                  // 32 x 64 x 2 workgroups
  dim3 block(NTHR);                           // 4 waves (wave32)
  esim_attn_kernel<<<grid, block, 0, stream>>>(ctx, rsp, cm, rm, out);
}
